// Gophormer_62612033241329
// MI455X (gfx1250) — matrix operand order repaired and checked by reading, not compiled, not run on hardware
//
#include <hip/hip_runtime.h>
#include <hip/hip_bf16.h>

#define N_NODES 4096
#define E_EDGES 131072
#define IN_DIM  128
#define OUT_DIM 64
#define DMODEL  512
#define DFF     2048
#define NHEAD   8
#define DHEAD   64
#define SQ      4097      /* N + G tokens */
#define SPAD    4160      /* SQ padded to multiple of 64 */
#define WORDS   130       /* ceil(SPAD/32) mask words per row */

typedef unsigned short u16;
typedef unsigned int   u32;
typedef __attribute__((ext_vector_type(16))) __bf16 v16bf;
typedef __attribute__((ext_vector_type(8)))  float  v8f;
typedef __attribute__((ext_vector_type(4)))  u32    u32x4;
typedef __attribute__((ext_vector_type(4)))  float  f32x4;
typedef __attribute__((ext_vector_type(4)))  u16    u16x4;

union Frag { u32x4 q[2]; u32 u[8]; v16bf v; };

__device__ __forceinline__ u16 f2bf(float f) {
  u32 u = __builtin_bit_cast(u32, f);
  u32 r = u + 0x7fffu + ((u >> 16) & 1u);
  return (u16)(r >> 16);
}

__device__ __forceinline__ v8f wmma_bf16(v16bf a, v16bf b, v8f c) {
  return __builtin_amdgcn_wmma_f32_16x16x32_bf16(false, a, false, b, (short)0, c,
                                                 false, false);
}

// A-fragment (16x32 bf16), CDNA5 layout: lane<16: row=lane, K in {0..7,16..23};
// lane>=16: row=lane-16, K in {8..15,24..31}. rowD = 16B-aligned dword ptr to row.
__device__ __forceinline__ v16bf afrag(const u32* rowD, int lane) {
  Frag f; const int k0 = (lane < 16) ? 0 : 4;
  f.q[0] = *(const u32x4*)(rowD + k0);
  f.q[1] = *(const u32x4*)(rowD + k0 + 8);
  return f.v;
}

// B-fragment (32x16 bf16): lane<16: col=lane, K in {0..7,16..23}; lane>=16: col=lane-16, K in {8..15,24..31}.
// colD = 16B-aligned dword ptr to this lane's column (K contiguous).
__device__ __forceinline__ v16bf bfrag(const u32* colD, int lane) {
  Frag f; const int k0 = (lane < 16) ? 0 : 4;
  f.q[0] = *(const u32x4*)(colD + k0);
  f.q[1] = *(const u32x4*)(colD + k0 + 8);
  return f.v;
}

// ---------------------------------------------------------------- mask / degree
__global__ __launch_bounds__(256) void k_mask_row0(u32* mask) {
  int w = blockIdx.x * 256 + threadIdx.x;
  if (w < WORDS) {
    int rem = SQ - w * 32;
    mask[w] = rem >= 32 ? 0xffffffffu : (rem <= 0 ? 0u : ((1u << rem) - 1u));
  }
}

__global__ __launch_bounds__(256) void k_mask_node(u32* mask) {
  int i = blockIdx.x * 256 + threadIdx.x;
  if (i < N_NODES) {
    int r = i + 1;
    u32* row = mask + (size_t)r * WORDS;
    int w = r >> 5;
    if (w == 0) row[0] = 1u | (1u << (r & 31));
    else { row[0] = 1u; row[w] = 1u << (r & 31); }
  }
}

__global__ __launch_bounds__(256) void k_mask_edges(const long long* __restrict__ ei,
                                                    u32* mask) {
  int e = blockIdx.x * 256 + threadIdx.x;
  if (e < E_EDGES) {
    int r = (int)ei[e] + 1;
    int c = (int)ei[E_EDGES + e] + 1;
    atomicOr(&mask[(size_t)r * WORDS + (c >> 5)], 1u << (c & 31));
  }
}

__global__ __launch_bounds__(256) void k_deg(const u32* __restrict__ mask, float* deg) {
  int i = blockIdx.x * 256 + threadIdx.x;
  if (i < N_NODES) {
    const u32* row = mask + (size_t)(i + 1) * WORDS;
    int c = 0;
#pragma unroll 10
    for (int w = 0; w < WORDS; ++w) c += __popc(row[w]);
    deg[i] = (float)(c - 1);   // subtract always-set global-token column bit
  }
}

__global__ __launch_bounds__(256) void k_norm(const float* __restrict__ deg, float* nrm) {
  __shared__ float s[256];
  float a = 0.f;
  for (int i = threadIdx.x; i < N_NODES; i += 256) { float d = deg[i]; a += d * d; }
  s[threadIdx.x] = a; __syncthreads();
  for (int off = 128; off; off >>= 1) {
    if (threadIdx.x < off) s[threadIdx.x] += s[threadIdx.x + off];
    __syncthreads();
  }
  if (threadIdx.x == 0) nrm[0] = sqrtf(s[0]);
}

__global__ __launch_bounds__(256) void k_anchor_p(const float* __restrict__ deg,
                                                  const float* __restrict__ nrm,
                                                  float* __restrict__ p) {
  int i = blockIdx.x * 256 + threadIdx.x;
  if (i < N_NODES) {
    float inv = 1.f / (nrm[0] + 1e-6f);
    float s[4], mx = -3e30f;
#pragma unroll
    for (int a = 0; a < 4; ++a) { s[a] = deg[i] * deg[a] * inv; mx = fmaxf(mx, s[a]); }
    float sum = 0.f;
#pragma unroll
    for (int a = 0; a < 4; ++a) { s[a] = __expf(s[a] - mx); sum += s[a]; }
    float r = 1.f / sum;
#pragma unroll
    for (int a = 0; a < 4; ++a) p[i * 4 + a] = s[a] * r;
  }
}

// t_f32 init: row0 = global token; rows 1..SQ-1 = APE; pad rows = 0
__global__ __launch_bounds__(256) void k_tinit(const float* __restrict__ p,
                                               const float* __restrict__ anchor,
                                               const float* __restrict__ gtok,
                                               float* __restrict__ tf) {
  int idx = blockIdx.x * 256 + threadIdx.x;
  if (idx >= SPAD * DMODEL) return;
  int row = idx >> 9, col = idx & (DMODEL - 1);
  float v;
  if (row == 0) v = gtok[col];
  else if (row < SQ) {
    int i = row - 1;
    v = p[i * 4 + 0] * anchor[col] + p[i * 4 + 1] * anchor[DMODEL + col] +
        p[i * 4 + 2] * anchor[2 * DMODEL + col] + p[i * 4 + 3] * anchor[3 * DMODEL + col];
  } else v = 0.f;
  tf[idx] = v;
}

// float4 -> 4x bf16 (row-major copy, n multiple of 4)
__global__ __launch_bounds__(256) void k_cvt_bf16(const float* __restrict__ s,
                                                  u16* __restrict__ d, int n4) {
  int i = blockIdx.x * 256 + threadIdx.x;
  if (i < n4) {
    f32x4 v = ((const f32x4*)s)[i];
    u16x4 o;
#pragma unroll
    for (int j = 0; j < 4; ++j) o[j] = f2bf(v[j]);
    ((u16x4*)d)[i] = o;
  }
}

// f32 [K][N] row-major -> bf16 [N][K] column-major (weights, once per call)
__global__ __launch_bounds__(256) void k_cvt_bf16_t(const float* __restrict__ s,
                                                    u16* __restrict__ d,
                                                    int Kd, int Nd) {
  int i = blockIdx.x * 256 + threadIdx.x;
  if (i < Kd * Nd) {
    int k = i / Nd, n = i - k * Nd;
    d[(size_t)n * Kd + k] = f2bf(s[i]);
  }
}

// ---------------------------------------------------------------- tiled bf16 GEMM
// C[M,N] = A[M,K](bf16 row-major) @ B[K,N] (given as Bt[N][K] bf16)
// epilogues: +bias (valid A rows), +resid, relu -> f32 / bf16 / bf16-transposed
template<int N, int K>
__global__ __launch_bounds__(256)
void k_gemm_bf16(const u16* __restrict__ A, const u16* __restrict__ Bt,
                 const float* __restrict__ bias, const float* __restrict__ resid,
                 float* __restrict__ Cf, u16* __restrict__ Cb, u16* __restrict__ CbT,
                 int ldT, int arow_off, int arow_cnt, int relu) {
  __shared__ __align__(16) u16 As[2][64][40];  // [M][K], row stride 80B (bank-safe)
  __shared__ __align__(16) u16 Bs[2][64][40];  // [N][K]
  const int tid = threadIdx.x, lane = tid & 31, w = tid >> 5;
  const int wr = w & 3, wc = w >> 2;
  const int m0 = blockIdx.y * 64, n0 = blockIdx.x * 64;

  // cooperative tile loads: one 16B chunk per thread per matrix per K-step
  const int lm  = tid >> 2;            // row (A) / col (Bt): 0..63
  const int seg = tid & 3;             // which 16B of the 64B K-slice
  const int ar  = m0 + lm + arow_off;
  const bool av = (ar >= 0 && ar < arow_cnt);
  const u16* Arow = A  + (size_t)(av ? ar : 0) * K + seg * 8;
  const u16* Brow = Bt + (size_t)(n0 + lm) * K + seg * 8;
  const u32x4 zero = {0u, 0u, 0u, 0u};

  v8f c0 = {}, c1 = {};
  constexpr int NK = K / 32;
  u32x4 ra = *(const u32x4*)(Arow);
  u32x4 rb = *(const u32x4*)(Brow);
  if (!av) ra = zero;

#pragma unroll 2
  for (int kt = 0; kt < NK; ++kt) {
    const int cur = kt & 1;
    *(u32x4*)&As[cur][lm][seg * 8] = ra;   // ds_store_b128
    *(u32x4*)&Bs[cur][lm][seg * 8] = rb;
    __syncthreads();
    if (kt + 1 < NK) {                     // overlap next global loads with WMMA
      ra = *(const u32x4*)(Arow + (kt + 1) * 32);
      rb = *(const u32x4*)(Brow + (kt + 1) * 32);
      if (!av) ra = zero;
      if (kt + 2 < NK) {
        __builtin_prefetch(Arow + (kt + 2) * 32, 0, 1);
        __builtin_prefetch(Brow + (kt + 2) * 32, 0, 1);
      }
    }
    const u32* AsD = (const u32*)&As[cur][0][0];
    const u32* BsD = (const u32*)&Bs[cur][0][0];
    v16bf a  = afrag(&AsD[(wr * 16 + (lane & 15)) * 20], lane);
    v16bf b0 = bfrag(&BsD[(wc * 32 + (lane & 15)) * 20], lane);
    v16bf b1 = bfrag(&BsD[(wc * 32 + 16 + (lane & 15)) * 20], lane);
    c0 = wmma_bf16(a, b0, c0);
    c1 = wmma_bf16(a, b1, c1);
  }

#pragma unroll
  for (int t = 0; t < 2; ++t) {
    v8f c = t ? c1 : c0;
#pragma unroll
    for (int r = 0; r < 8; ++r) {
      int row = m0 + wr * 16 + ((lane < 16) ? r : r + 8);
      int col = n0 + wc * 32 + t * 16 + (lane & 15);
      float val = c[r];
      int arr = row + arow_off;
      bool arv = (arr >= 0 && arr < arow_cnt);
      if (bias && arv) val += bias[col];
      if (resid) val += resid[(size_t)row * N + col];
      if (relu) val = fmaxf(val, 0.f);
      if (Cf) Cf[(size_t)row * N + col] = val;
      if (Cb) Cb[(size_t)row * N + col] = f2bf(val);
      if (CbT) CbT[(size_t)col * ldT + row] = f2bf(val);
    }
  }
}

// ---------------------------------------------------------------- flash attention
// 1 wave per (head, 16-row q tile); 32-key blocks; skips fully-masked blocks.
__global__ __launch_bounds__(128)
void k_flash(const u16* __restrict__ qb, const u16* __restrict__ kb,
             const u16* __restrict__ vT, const u32* __restrict__ mask,
             u16* __restrict__ ctx) {
  __shared__ __align__(16) u16 Pls[4][16][40];     // per-wave P staging
  const int lane = threadIdx.x & 31, w = threadIdx.x >> 5;
  const int qt = blockIdx.x * 4 + w;
  if (qt >= (SQ + 15) / 16) return;
  const int h = blockIdx.y;
  const int q0 = qt * 16;
  const int qrow = q0 + (lane & 15);

  v16bf aq0 = afrag((const u32*)(qb + (size_t)qrow * DMODEL + h * DHEAD), lane);
  v16bf aq1 = afrag((const u32*)(qb + (size_t)qrow * DMODEL + h * DHEAD + 32), lane);

  float m[8], l[8];
#pragma unroll
  for (int r = 0; r < 8; ++r) { m[r] = -3e30f; l[r] = 0.f; }
  v8f o0 = {}, o1 = {}, o2 = {}, o3 = {};
  u32* PD = (u32*)&Pls[w][0][0];                    // row stride 20 dwords

  for (int kbI = 0; kbI < WORDS; ++kbI) {
    const int u0 = kbI * 32;
    u32 mw[8], ok = 0;
#pragma unroll
    for (int r = 0; r < 8; ++r) {
      int row = q0 + ((lane < 16) ? r : r + 8);
      mw[r] = mask[(size_t)row * WORDS + kbI];
      ok |= mw[r];
    }
    u32 any = ok;
    any |= __shfl_xor(any, 16, 32); any |= __shfl_xor(any, 8, 32);
    any |= __shfl_xor(any, 4, 32);  any |= __shfl_xor(any, 2, 32);
    any |= __shfl_xor(any, 1, 32);
    if (kbI && !any) continue;                      // block fully masked -> skip

    // scores S[16x32] = Q(16x64) . K^T
    const u16* kp0 = kb + (size_t)(u0 + (lane & 15)) * DMODEL + h * DHEAD;
    const u16* kp1 = kb + (size_t)(u0 + 16 + (lane & 15)) * DMODEL + h * DHEAD;
    v8f s0 = {}, s1 = {};
    s0 = wmma_bf16(aq0, bfrag((const u32*)kp0, lane), s0);
    s0 = wmma_bf16(aq1, bfrag((const u32*)(kp0 + 32), lane), s0);
    s1 = wmma_bf16(aq0, bfrag((const u32*)kp1, lane), s1);
    s1 = wmma_bf16(aq1, bfrag((const u32*)(kp1 + 32), lane), s1);

    // masked online softmax (row reductions across 16-lane halves)
#pragma unroll
    for (int r = 0; r < 8; ++r) {
      float a0 = ((mw[r] >> (lane & 15)) & 1u)        ? s0[r] * 0.125f : -1e9f;
      float a1 = ((mw[r] >> (16 + (lane & 15))) & 1u) ? s1[r] * 0.125f : -1e9f;
      float rm = fmaxf(a0, a1);
      rm = fmaxf(rm, __shfl_xor(rm, 1, 32)); rm = fmaxf(rm, __shfl_xor(rm, 2, 32));
      rm = fmaxf(rm, __shfl_xor(rm, 4, 32)); rm = fmaxf(rm, __shfl_xor(rm, 8, 32));
      float mn = fmaxf(m[r], rm);
      float sc = __expf(m[r] - mn);
      float p0 = __expf(a0 - mn), p1 = __expf(a1 - mn);
      float rs = p0 + p1;
      rs += __shfl_xor(rs, 1, 32); rs += __shfl_xor(rs, 2, 32);
      rs += __shfl_xor(rs, 4, 32); rs += __shfl_xor(rs, 8, 32);
      l[r] = l[r] * sc + rs;
      m[r] = mn;
      o0[r] *= sc; o1[r] *= sc; o2[r] *= sc; o3[r] *= sc;
      int rloc = (lane < 16) ? r : r + 8;
      Pls[w][rloc][(lane & 15)]      = f2bf(p0);
      Pls[w][rloc][16 + (lane & 15)] = f2bf(p1);
    }
    __builtin_amdgcn_fence(__ATOMIC_RELEASE, "wavefront");
    __builtin_amdgcn_wave_barrier();
    __builtin_amdgcn_fence(__ATOMIC_ACQUIRE, "wavefront");

    // O += P(16x32) . V(32x64), V pre-transposed [D][SPAD]
    v16bf ap = afrag(&PD[(lane & 15) * 20], lane);
    const u16* vp = vT + (size_t)(h * DHEAD + (lane & 15)) * SPAD + u0;
    o0 = wmma_bf16(ap, bfrag((const u32*)vp, lane), o0);
    o1 = wmma_bf16(ap, bfrag((const u32*)(vp + 16 * SPAD), lane), o1);
    o2 = wmma_bf16(ap, bfrag((const u32*)(vp + 32 * SPAD), lane), o2);
    o3 = wmma_bf16(ap, bfrag((const u32*)(vp + 48 * SPAD), lane), o3);
    __builtin_amdgcn_wave_barrier();
  }

#pragma unroll
  for (int r = 0; r < 8; ++r) {
    int row = q0 + ((lane < 16) ? r : r + 8);
    if (row < SQ) {
      float inv = 1.f / l[r];
      size_t base = (size_t)row * DMODEL + h * DHEAD + (lane & 15);
      ctx[base]      = f2bf(o0[r] * inv);
      ctx[base + 16] = f2bf(o1[r] * inv);
      ctx[base + 32] = f2bf(o2[r] * inv);
      ctx[base + 48] = f2bf(o3[r] * inv);
    }
  }
}

// ---------------------------------------------------------------- layernorm
__global__ __launch_bounds__(256)
void k_ln(const float* __restrict__ in, const float* __restrict__ g,
          const float* __restrict__ b, float* __restrict__ tf, u16* __restrict__ tb) {
  __shared__ float s1[256], s2[256];
  const int tid = threadIdx.x;
  const size_t base = (size_t)blockIdx.x * DMODEL;
  float a0 = in[base + tid], a1 = in[base + tid + 256];
  s1[tid] = a0 + a1; s2[tid] = a0 * a0 + a1 * a1;
  __syncthreads();
  for (int off = 128; off; off >>= 1) {
    if (tid < off) { s1[tid] += s1[tid + off]; s2[tid] += s2[tid + off]; }
    __syncthreads();
  }
  float mu = s1[0] * (1.f / DMODEL);
  float var = s2[0] * (1.f / DMODEL) - mu * mu;
  float rs = rsqrtf(var + 1e-5f);
  float y0 = (a0 - mu) * rs * g[tid] + b[tid];
  float y1 = (a1 - mu) * rs * g[tid + 256] + b[tid + 256];
  tf[base + tid] = y0;       tf[base + tid + 256] = y1;
  tb[base + tid] = f2bf(y0); tb[base + tid + 256] = f2bf(y1);
}

// ---------------------------------------------------------------- log_softmax (64 cols)
__global__ __launch_bounds__(128)
void k_logsoftmax(const float* __restrict__ logits, float* __restrict__ out) {
  const int lane = threadIdx.x & 31, w = threadIdx.x >> 5;
  const int row = blockIdx.x * 4 + w;
  const float* x = logits + (size_t)row * OUT_DIM;
  float a0 = x[lane], a1 = x[lane + 32];
  float mx = fmaxf(a0, a1);
  mx = fmaxf(mx, __shfl_xor(mx, 16, 32)); mx = fmaxf(mx, __shfl_xor(mx, 8, 32));
  mx = fmaxf(mx, __shfl_xor(mx, 4, 32));  mx = fmaxf(mx, __shfl_xor(mx, 2, 32));
  mx = fmaxf(mx, __shfl_xor(mx, 1, 32));
  float s = __expf(a0 - mx) + __expf(a1 - mx);
  s += __shfl_xor(s, 16, 32); s += __shfl_xor(s, 8, 32);
  s += __shfl_xor(s, 4, 32);  s += __shfl_xor(s, 2, 32); s += __shfl_xor(s, 1, 32);
  float lg = logf(s);
  out[(size_t)row * OUT_DIM + lane]      = a0 - mx - lg;
  out[(size_t)row * OUT_DIM + lane + 32] = a1 - mx - lg;
}

// ================================================================ host side
extern "C" void kernel_launch(void* const* d_in, const int* in_sizes, int n_in,
                              void* d_out, int out_size, void* d_ws, size_t ws_size,
                              hipStream_t stream) {
  (void)in_sizes; (void)n_in; (void)out_size; (void)ws_size;
  // setup_inputs() dict order (insertion order, nested):
  const float*     x      = (const float*)d_in[0];
  const long long* ei     = (const long long*)d_in[1];
  const float*     emb_W  = (const float*)d_in[2];
  const float*     emb_b  = (const float*)d_in[3];
  const float*     anchor = (const float*)d_in[4];
  const float*     gtok   = (const float*)d_in[5];
  const float*     out_W  = (const float*)d_in[38];
  const float*     out_b  = (const float*)d_in[39];

  char* ws = (char*)d_ws; size_t off = 0;
  auto alloc = [&](size_t bytes) -> void* {
    void* p = ws + off; off = (off + bytes + 255) & ~(size_t)255; return p;
  };
  u32*   mask   = (u32*)  alloc((size_t)SPAD * WORDS * 4);
  float* deg    = (float*)alloc(N_NODES * 4);
  float* nrm    = (float*)alloc(256);
  float* pA     = (float*)alloc(N_NODES * 4 * 4);
  float* tf     = (float*)alloc((size_t)SPAD * DMODEL * 4);
  float* resid  = (float*)alloc((size_t)SPAD * DMODEL * 4);
  u16*   tb     = (u16*)  alloc((size_t)SPAD * DMODEL * 2);
  u16*   qbuf   = (u16*)  alloc((size_t)SPAD * DMODEL * 2);
  u16*   kbuf   = (u16*)  alloc((size_t)SPAD * DMODEL * 2);
  u16*   vTbuf  = (u16*)  alloc((size_t)DMODEL * SPAD * 2);
  u16*   ctxb   = (u16*)  alloc((size_t)SPAD * DMODEL * 2);
  u16*   h1b    = (u16*)  alloc((size_t)SPAD * DFF * 2);
  u16*   xb     = (u16*)  alloc((size_t)N_NODES * IN_DIM * 2);
  float* logits = (float*)alloc((size_t)N_NODES * OUT_DIM * 4);
  u16*   embWb  = (u16*)  alloc((size_t)IN_DIM * DMODEL * 2);
  u16*   outWb  = (u16*)  alloc((size_t)DMODEL * OUT_DIM * 2);
  u16 *Wqb[2], *Wkb[2], *Wvb[2], *Wob[2], *W1b[2], *W2b[2];
  for (int l = 0; l < 2; ++l) {
    Wqb[l] = (u16*)alloc((size_t)DMODEL * DMODEL * 2);
    Wkb[l] = (u16*)alloc((size_t)DMODEL * DMODEL * 2);
    Wvb[l] = (u16*)alloc((size_t)DMODEL * DMODEL * 2);
    Wob[l] = (u16*)alloc((size_t)DMODEL * DMODEL * 2);
    W1b[l] = (u16*)alloc((size_t)DMODEL * DFF * 2);
    W2b[l] = (u16*)alloc((size_t)DFF * DMODEL * 2);
  }

  auto cvtT = [&](const float* s, u16* d, int K, int N) {
    k_cvt_bf16_t<<<(K * N + 255) / 256, 256, 0, stream>>>(s, d, K, N);
  };

  // --- adjacency / attention mask, degrees, anchor encoding
  hipMemsetAsync(mask, 0, (size_t)SPAD * WORDS * 4, stream);
  k_mask_row0 <<<(WORDS + 255) / 256, 256, 0, stream>>>(mask);
  k_mask_node <<<N_NODES / 256, 256, 0, stream>>>(mask);
  k_mask_edges<<<E_EDGES / 256, 256, 0, stream>>>(ei, mask);
  k_deg       <<<N_NODES / 256, 256, 0, stream>>>(mask, deg);
  k_norm      <<<1, 256, 0, stream>>>(deg, nrm);
  k_anchor_p  <<<N_NODES / 256, 256, 0, stream>>>(deg, nrm, pA);
  k_tinit     <<<(SPAD * DMODEL) / 256, 256, 0, stream>>>(pA, anchor, gtok, tf);

  // --- bf16 conversions: A-side row-major, weights transposed to [N][K]
  k_cvt_bf16<<<(N_NODES * IN_DIM / 4 + 255) / 256, 256, 0, stream>>>(
      x, xb, N_NODES * IN_DIM / 4);
  cvtT(emb_W, embWb, IN_DIM, DMODEL);
  cvtT(out_W, outWb, DMODEL, OUT_DIM);
  for (int l = 0; l < 2; ++l) {
    int b = 6 + l * 16;
    cvtT((const float*)d_in[b + 0],  Wqb[l], DMODEL, DMODEL);
    cvtT((const float*)d_in[b + 2],  Wkb[l], DMODEL, DMODEL);
    cvtT((const float*)d_in[b + 4],  Wvb[l], DMODEL, DMODEL);
    cvtT((const float*)d_in[b + 6],  Wob[l], DMODEL, DMODEL);
    cvtT((const float*)d_in[b + 12], W1b[l], DMODEL, DFF);
    cvtT((const float*)d_in[b + 14], W2b[l], DFF, DMODEL);
  }

  // --- embedding: t = x@emb_W + emb_b + (ape|token), A row = t row - 1
  k_gemm_bf16<DMODEL, IN_DIM><<<dim3(DMODEL / 64, SPAD / 64), 256, 0, stream>>>(
      xb, embWb, emb_b, tf, tf, tb, nullptr, 0, -1, N_NODES, 0);

  // --- transformer layers
  for (int l = 0; l < 2; ++l) {
    int b = 6 + l * 16;
    const float* bq  = (const float*)d_in[b + 1];
    const float* bk  = (const float*)d_in[b + 3];
    const float* bv  = (const float*)d_in[b + 5];
    const float* bo  = (const float*)d_in[b + 7];
    const float* g1  = (const float*)d_in[b + 8];
    const float* be1 = (const float*)d_in[b + 9];
    const float* g2  = (const float*)d_in[b + 10];
    const float* be2 = (const float*)d_in[b + 11];
    const float* b1  = (const float*)d_in[b + 13];
    const float* b2  = (const float*)d_in[b + 15];

    k_gemm_bf16<DMODEL, DMODEL><<<dim3(8, SPAD / 64), 256, 0, stream>>>(
        tb, Wqb[l], bq, nullptr, nullptr, qbuf, nullptr, 0, 0, SQ, 0);
    k_gemm_bf16<DMODEL, DMODEL><<<dim3(8, SPAD / 64), 256, 0, stream>>>(
        tb, Wkb[l], bk, nullptr, nullptr, kbuf, nullptr, 0, 0, SQ, 0);
    k_gemm_bf16<DMODEL, DMODEL><<<dim3(8, SPAD / 64), 256, 0, stream>>>(
        tb, Wvb[l], bv, nullptr, nullptr, nullptr, vTbuf, SPAD, 0, SQ, 0);

    dim3 fg(((SQ + 15) / 16 + 3) / 4, NHEAD);
    k_flash<<<fg, 128, 0, stream>>>(qbuf, kbuf, vTbuf, mask, ctxb);

    k_gemm_bf16<DMODEL, DMODEL><<<dim3(8, SPAD / 64), 256, 0, stream>>>(
        ctxb, Wob[l], bo, tf, resid, nullptr, nullptr, 0, 0, SQ, 0);
    k_ln<<<SPAD, 256, 0, stream>>>(resid, g1, be1, tf, tb);

    k_gemm_bf16<DFF, DMODEL><<<dim3(DFF / 64, SPAD / 64), 256, 0, stream>>>(
        tb, W1b[l], b1, nullptr, nullptr, h1b, nullptr, 0, 0, SQ, 1);
    k_gemm_bf16<DMODEL, DFF><<<dim3(8, SPAD / 64), 256, 0, stream>>>(
        h1b, W2b[l], b2, tf, resid, nullptr, nullptr, 0, 0, SQ, 0);
    k_ln<<<SPAD, 256, 0, stream>>>(resid, g2, be2, tf, tb);
  }

  // --- output head: logits = t[1:] @ out_W + out_b ; log_softmax
  k_gemm_bf16<OUT_DIM, DMODEL><<<dim3(1, N_NODES / 64), 256, 0, stream>>>(
      tb, outWb, out_b, nullptr, logits, nullptr, nullptr, 0, +1, SQ, 0);
  k_logsoftmax<<<N_NODES / 4, 128, 0, stream>>>(logits, (float*)d_out);
}
